// Encoder_9981503996219
// MI455X (gfx1250) — matrix operand order repaired and checked by reading, not compiled, not run on hardware
//
#include <hip/hip_runtime.h>
#include <hip/hip_bf16.h>
#include <cstdint>
#include <cstddef>

typedef __attribute__((ext_vector_type(16))) _Float16 v16h;
typedef __attribute__((ext_vector_type(8)))  _Float16 v8h;
typedef __attribute__((ext_vector_type(8)))  float    v8f;

#define HID   128
#define GATES 512
#define BAT   16
#define SEQL  361
#define NFEAT 49

// workspace byte offsets
#define O_FIRST  0u
#define O_FLAGS  4096u
#define O_X      8192u
#define O_WHH    (2u<<20)     // 4*512*128 f16
#define O_WIH1   (3u<<20)     // 2*512*256 f16
#define O_Y0     (4u<<20)     // 2*361*16*256 f16 (double buffer)
#define O_XG     (12u<<20)    // 2 * 361*8*4*256 f32, fragment layout

__device__ __forceinline__ v16h cat8(v8h a, v8h b) {
  return __builtin_shufflevector(a, b, 0,1,2,3,4,5,6,7,8,9,10,11,12,13,14,15);
}
// A fragment (16x32 f16, row-major a[m][ld]): lane m=L&15, hi=L>>4
__device__ __forceinline__ v16h loadA(const _Float16* a, int ld, int k0, int m, int hi) {
  const _Float16* p = a + m * ld + k0 + hi * 16;
  return cat8(*(const v8h*)p, *(const v8h*)(p + 8));
}
// B fragment (32x16 f16) from row-major W[n][ld]: lane n=L&15 holds 16 contiguous K
__device__ __forceinline__ v16h loadB(const _Float16* w, int ld, int nrow, int k0) {
  const _Float16* p = w + (size_t)nrow * ld + k0;
  return cat8(*(const v8h*)p, *(const v8h*)(p + 8));
}
__device__ __forceinline__ v8f wmma16(v16h a, v16h b, v8f c) {
  return __builtin_amdgcn_wmma_f32_16x16x32_f16(false, a, false, b, (short)0, c, false, false);
}
__device__ __forceinline__ v8f zero8() {
  v8f z = {0.f,0.f,0.f,0.f,0.f,0.f,0.f,0.f};
  return z;
}

// gfx1250 has a hardware TANH transcendental; use it if the builtin exists.
#if __has_builtin(__builtin_amdgcn_tanhf)
__device__ __forceinline__ float tanh_hw(float x) { return __builtin_amdgcn_tanhf(x); }
#elif __has_builtin(__builtin_amdgcn_tanh_f32)
__device__ __forceinline__ float tanh_hw(float x) { return __builtin_amdgcn_tanh_f32(x); }
#else
__device__ __forceinline__ float tanh_hw(float x) {
  x = fminf(fmaxf(x, -15.f), 15.f);
  float e = __expf(2.f * x);
  return (e - 1.f) / (e + 1.f);
}
#endif
__device__ __forceinline__ float sigt(float x) { return 0.5f + 0.5f * tanh_hw(0.5f * x); }

// ---- prep: firstIdx[p] = min flat index (f*16+b) with data[b,f]==p; zero sync flags
__global__ void k_prep(const int* __restrict__ data, unsigned* __restrict__ first,
                       unsigned* __restrict__ flags) {
  int tid = threadIdx.x;
  for (int i = tid; i < SEQL; i += 512) first[i] = 0x7fffffffu;
  for (int i = tid; i < 128; i += 512) flags[i] = 0u;
  __threadfence();
  __syncthreads();
  for (int i = tid; i < 48 * BAT; i += 512) {
    int f = i >> 4, b = i & 15;
    int v = data[b * NFEAT + f];
    if (v > 0 && v < SEQL) atomicMin(&first[v], (unsigned)i);
  }
}

// ---- board tensor, transposed for broadcast reads: X[f][t][b]
__global__ void k_buildX(const int* __restrict__ data, const unsigned* __restrict__ first,
                         float* __restrict__ X) {
  int e = blockIdx.x * blockDim.x + threadIdx.x;
  if (e >= NFEAT * BAT * SEQL) return;
  int b = e & 15;
  int t = (e >> 4) % SEQL;
  int f = e / (SEQL * BAT);
  float v;
  if (f < 48) v = (t > 0 && first[t] <= (unsigned)(f * 16 + b)) ? 1.f : 0.f;
  else        v = (t == 0) ? (float)data[b * NFEAT + 48] : ((first[t] <= 767u) ? 1.f : 0.f);
  X[((size_t)f * SEQL + t) * BAT + b] = v;
}

// ---- fp32 -> f16 weight conversion (row-major [n][k] == WMMA-B friendly)
__global__ void k_cvt(const float* __restrict__ whh0, const float* __restrict__ whh1,
                      const float* __restrict__ wih1,
                      _Float16* __restrict__ whh16, _Float16* __restrict__ wih116) {
  int i = blockIdx.x * blockDim.x + threadIdx.x;
  if (i < 4 * GATES * HID) {
    int role = i / (GATES * HID), rem = i % (GATES * HID);
    float s = (role < 2) ? whh0[role * GATES * HID + rem] : whh1[(role - 2) * GATES * HID + rem];
    whh16[i] = (_Float16)s;
  }
  if (i < 2 * GATES * 256) wih116[i] = (_Float16)wih1[i];
}

// ---- persistent LSTM: 4 WGs = {l0 fwd, l0 bwd, l1 fwd, l1 bwd}, feat-pipelined.
// Wave w owns gate tiles {w, w+8, w+16, w+24}: acc[j] = gate j (i,f,g,o) for
// hidden n in [16w,16w+16)  ->  cell update is register-only.
__launch_bounds__(256, 1)
__global__ void k_lstm(const float* __restrict__ X,
                       const _Float16* __restrict__ whh16,
                       const _Float16* __restrict__ wih116,
                       const float* __restrict__ wih0,
                       const float* __restrict__ bih0, const float* __restrict__ bhh0,
                       const float* __restrict__ bih1, const float* __restrict__ bhh1,
                       const float* __restrict__ h0, const float* __restrict__ c0,
                       _Float16* __restrict__ y0, float* __restrict__ xg,
                       float* __restrict__ out, unsigned* __restrict__ flags) {
  const int role = blockIdx.x, layer = role >> 1, dir = role & 1;
  const int tid = threadIdx.x, lane = tid & 31, wave = tid >> 5;
  const int nl = lane & 15, hl = lane >> 4;
  const int ng = wave * 16 + nl;            // hidden index handled in cell update

  __shared__ _Float16 hA[2][BAT * 136];     // double-buffered h (f16, row pad 136)
  __shared__ float    Xs[SEQL * BAT];       // layer0: per-feat board slice

  // per-lane bias / layer0 input column (4 gates each), c-state in registers
  float biasR[4], wxR[4];
  {
    const float* bi = (layer == 0) ? bih0 : bih1;
    const float* bh = (layer == 0) ? bhh0 : bhh1;
#pragma unroll
    for (int j = 0; j < 4; ++j) {
      biasR[j] = bi[dir * GATES + j * HID + ng] + bh[dir * GATES + j * HID + ng];
      wxR[j] = (layer == 0) ? wih0[dir * GATES + j * HID + ng] : 0.f;
    }
  }
  v8f cReg;
#pragma unroll
  for (int r = 0; r < 8; ++r) cReg[r] = c0[role * 2048 + (r + 8 * hl) * HID + ng];
  for (int i = tid; i < BAT * HID; i += 256) {
    int m = i >> 7, k = i & 127;
    hA[0][m * 136 + k] = (_Float16)h0[role * 2048 + i];
  }
  __syncthreads();

  unsigned* flag0 = flags;        // layer0 done-count per feat
  unsigned* flag1 = flags + 64;   // layer1 done-count per feat
  const _Float16* wb = whh16 + (size_t)role * (GATES * HID);
  float* xgd = xg + (size_t)dir * ((size_t)SEQL * 8 * 4 * 256);  // fragment layout

  // recurrent weight fragments resident in VGPRs for the whole kernel
  v16h Bf[4][4];
#pragma unroll
  for (int j = 0; j < 4; ++j)
#pragma unroll
    for (int s = 0; s < 4; ++s)
      Bf[j][s] = loadB(wb, HID, (wave + 8 * j) * 16 + nl, s * 32 + hl * 16);

  int ph = 0;   // hA buffer currently holding valid h

  for (int f = 0; f < NFEAT; ++f) {
    const int parity = f & 1;
    _Float16* y0p = y0 + (size_t)parity * ((size_t)SEQL * BAT * 256);

    if (layer == 0) {
      if (f >= 2 && tid == 0) {   // don't overwrite y0 buffer layer1 hasn't consumed
        while (__hip_atomic_load(&flag1[f - 2], __ATOMIC_ACQUIRE, __HIP_MEMORY_SCOPE_AGENT) < 2u)
          __builtin_amdgcn_s_sleep(2);
      }
      // stage this feat's board slice into LDS (off the critical scan path)
      for (int i = tid; i < SEQL * BAT; i += 256)
        Xs[i] = X[(size_t)f * (SEQL * BAT) + i];
      __syncthreads();
    } else {
      if (tid == 0)
        while (__hip_atomic_load(&flag0[f], __ATOMIC_ACQUIRE, __HIP_MEMORY_SCOPE_AGENT) < 2u)
          __builtin_amdgcn_s_sleep(2);
      __syncthreads();
      // --- parallel input projection, stored directly in WMMA fragment layout:
      //     xgd[((t*8 + wave)*4 + j)*256 + lane*8 + r]
      const _Float16* wi = wih116 + (size_t)dir * (GATES * 256);
      for (int p2 = 0; p2 < 2; ++p2) {
        const int ja = p2 * 2;                       // gate pair (j, j+1)
        v16h Bp[2][8];                               // weights resident across all t
#pragma unroll
        for (int s = 0; s < 8; ++s) {
          Bp[0][s] = loadB(wi, 256, (wave + 8 * (ja    )) * 16 + nl, s * 32 + hl * 16);
          Bp[1][s] = loadB(wi, 256, (wave + 8 * (ja + 1)) * 16 + nl, s * 32 + hl * 16);
        }
        for (int t0 = 0; t0 + 4 <= SEQL; t0 += 4) {  // 90 full chunks, branch-free
          v8f acc[4][2];
#pragma unroll
          for (int q = 0; q < 4; ++q) { acc[q][0] = zero8(); acc[q][1] = zero8(); }
#pragma unroll
          for (int s = 0; s < 8; ++s) {
            v16h Aq[4];
#pragma unroll
            for (int q = 0; q < 4; ++q)
              Aq[q] = loadA(y0p + (size_t)(t0 + q) * (BAT * 256), 256, s * 32, nl, hl);
#pragma unroll
            for (int q = 0; q < 4; ++q) {
              acc[q][0] = wmma16(Aq[q], Bp[0][s], acc[q][0]);
              acc[q][1] = wmma16(Aq[q], Bp[1][s], acc[q][1]);
            }
          }
#pragma unroll
          for (int q = 0; q < 4; ++q) {
            *(v8f*)&xgd[(((size_t)(t0 + q) * 8 + wave) * 4 + ja    ) * 256 + lane * 8] = acc[q][0];
            *(v8f*)&xgd[(((size_t)(t0 + q) * 8 + wave) * 4 + ja + 1) * 256 + lane * 8] = acc[q][1];
          }
        }
        {  // tail t = 360
          v8f a0 = zero8(), a1 = zero8();
#pragma unroll
          for (int s = 0; s < 8; ++s) {
            v16h A = loadA(y0p + (size_t)(SEQL - 1) * (BAT * 256), 256, s * 32, nl, hl);
            a0 = wmma16(A, Bp[0][s], a0);
            a1 = wmma16(A, Bp[1][s], a1);
          }
          *(v8f*)&xgd[(((size_t)(SEQL - 1) * 8 + wave) * 4 + ja    ) * 256 + lane * 8] = a0;
          *(v8f*)&xgd[(((size_t)(SEQL - 1) * 8 + wave) * 4 + ja + 1) * 256 + lane * 8] = a1;
        }
      }
      __threadfence();
      __syncthreads();
    }

    for (int st = 0; st < SEQL; ++st) {
      const int t = dir ? (SEQL - 1 - st) : st;
      const _Float16* hr = hA[ph];
      _Float16*       hw = hA[ph ^ 1];
      // one LDS round trip: batch all 4 A fragments, then 16 back-to-back WMMAs
      v16h Af[4];
#pragma unroll
      for (int s = 0; s < 4; ++s) Af[s] = loadA(hr, 136, s * 32, nl, hl);
      v8f acc[4];
      if (layer == 0) {
#pragma unroll
        for (int j = 0; j < 4; ++j) acc[j] = zero8();
      } else {
#pragma unroll
        for (int j = 0; j < 4; ++j)
          acc[j] = *(const v8f*)&xgd[(((size_t)t * 8 + wave) * 4 + j) * 256 + lane * 8];
      }
#pragma unroll
      for (int s = 0; s < 4; ++s)
#pragma unroll
        for (int j = 0; j < 4; ++j) acc[j] = wmma16(Af[s], Bf[j][s], acc[j]);

      // register-only LSTM cell update: lane element r == (b=r+8*hl, n=ng)
      float xv[8];
      if (layer == 0) {
#pragma unroll
        for (int r = 0; r < 8; ++r) xv[r] = Xs[t * BAT + (r + 8 * hl)];
      }
#pragma unroll
      for (int r = 0; r < 8; ++r) {
        float gi = acc[0][r] + biasR[0];
        float gf = acc[1][r] + biasR[1];
        float gg = acc[2][r] + biasR[2];
        float go = acc[3][r] + biasR[3];
        if (layer == 0) {
          gi += xv[r] * wxR[0]; gf += xv[r] * wxR[1];
          gg += xv[r] * wxR[2]; go += xv[r] * wxR[3];
        }
        float c = cReg[r];
        c = sigt(gf) * c + sigt(gi) * tanh_hw(gg);
        float h = sigt(go) * tanh_hw(c);
        cReg[r] = c;
        hw[(r + 8 * hl) * 136 + ng] = (_Float16)h;
        if (layer == 0)
          y0p[((size_t)t * BAT + (r + 8 * hl)) * 256 + dir * HID + ng] = (_Float16)h;
        else
          out[((size_t)(r + 8 * hl) * NFEAT + f) * ((size_t)SEQL * 256)
              + (size_t)t * 256 + dir * HID + ng] = h;
      }
      ph ^= 1;
      __syncthreads();
    }

    if (tid == 0) {
      __threadfence();
      atomicAdd((layer == 0 ? flag0 : flag1) + f, 1u);
    }
    __syncthreads();
  }
}

extern "C" void kernel_launch(void* const* d_in, const int* in_sizes, int n_in,
                              void* d_out, int out_size, void* d_ws, size_t ws_size,
                              hipStream_t stream) {
  (void)in_sizes; (void)n_in; (void)out_size; (void)ws_size;
  const int*   data = (const int*)  d_in[0];
  const float* wih0 = (const float*)d_in[1];
  const float* whh0 = (const float*)d_in[2];
  const float* bih0 = (const float*)d_in[3];
  const float* bhh0 = (const float*)d_in[4];
  const float* wih1 = (const float*)d_in[5];
  const float* whh1 = (const float*)d_in[6];
  const float* bih1 = (const float*)d_in[7];
  const float* bhh1 = (const float*)d_in[8];
  const float* h0   = (const float*)d_in[9];
  const float* c0   = (const float*)d_in[10];

  char* ws = (char*)d_ws;
  unsigned*  first  = (unsigned*) (ws + O_FIRST);
  unsigned*  flags  = (unsigned*) (ws + O_FLAGS);
  float*     X      = (float*)    (ws + O_X);
  _Float16*  whh16  = (_Float16*) (ws + O_WHH);
  _Float16*  wih116 = (_Float16*) (ws + O_WIH1);
  _Float16*  y0     = (_Float16*) (ws + O_Y0);
  float*     xg     = (float*)    (ws + O_XG);
  float*     out    = (float*)d_out;

  k_prep<<<1, 512, 0, stream>>>(data, first, flags);
  int nX = NFEAT * BAT * SEQL;
  k_buildX<<<(nX + 255) / 256, 256, 0, stream>>>(data, first, X);
  k_cvt<<<(4 * GATES * HID + 255) / 256, 256, 0, stream>>>(whh0, whh1, wih1, whh16, wih116);
  k_lstm<<<4, 256, 0, stream>>>(X, whh16, wih116, wih0, bih0, bhh0, bih1, bhh1,
                                h0, c0, y0, xg, out, flags);
}
